// Malleblock_71476845740074
// MI455X (gfx1250) — matrix operand order repaired and checked by reading, not compiled, not run on hardware
//
#include <hip/hip_runtime.h>
#include <hip/hip_bf16.h>

typedef __attribute__((ext_vector_type(16))) __bf16 v16bf;
typedef __attribute__((ext_vector_type(8)))  __bf16 v8bf;
typedef __attribute__((ext_vector_type(8)))  float  v8f;

__device__ __forceinline__ __bf16 f2bf(float f) {
    unsigned u = __builtin_bit_cast(unsigned, f);
    unsigned r = u + 0x7FFFu + ((u >> 16) & 1u);   // round-to-nearest-even
    unsigned short hs = (unsigned short)(r >> 16);
    return __builtin_bit_cast(__bf16, hs);
}

__device__ __forceinline__ v16bf cat16(v8bf lo, v8bf hi) {
    return __builtin_shufflevector(lo, hi, 0, 1, 2, 3, 4, 5, 6, 7,
                                           8, 9, 10, 11, 12, 13, 14, 15);
}

__device__ __forceinline__ int refl(int i, int n) {
    if (i < 0) i = -i;
    if (i >= n) i = 2 * n - 2 - i;
    return i;
}

// ---------------------------------------------------------------------------
// x1 = x_[:, :, ::4, ::4]   (4,48,256,256) -> (4,48,64,64)
// ---------------------------------------------------------------------------
__global__ void downsample4_kernel(const float* __restrict__ src, float* __restrict__ dst) {
    int idx = blockIdx.x * 256 + threadIdx.x;          // 786432 total
    int x = idx & 63, y = (idx >> 6) & 63, nc = idx >> 12;
    dst[idx] = src[((size_t)nc << 16) + (size_t)(y << 2) * 256 + (x << 2)];
}

// ---------------------------------------------------------------------------
// 2x2 maxpool  (4,48,64,64) -> (4,48,32,32)
// ---------------------------------------------------------------------------
__global__ void maxpool2_kernel(const float* __restrict__ src, float* __restrict__ dst) {
    int idx = blockIdx.x * 256 + threadIdx.x;          // 196608 total
    int x = idx & 31, y = (idx >> 5) & 31, nc = idx >> 10;
    const float* s = src + ((size_t)nc << 12) + (size_t)(y << 1) * 64 + (x << 1);
    dst[idx] = fmaxf(fmaxf(s[0], s[1]), fmaxf(s[64], s[65]));
}

// ---------------------------------------------------------------------------
// 3x3 SAME conv, Cin=48 fixed, Cout<=48, via bf16 WMMA (im2col GEMM).
// M = Cout (NMT compile-time tiles of 16), K = 48*9 = 432 padded to 448,
// N = H*W.  Block = 128 threads = 4 waves; each wave owns 16 positions.
// ---------------------------------------------------------------------------
template <int NMT>
__global__ __launch_bounds__(128) void conv3x3_wmma(
    const float* __restrict__ in, const float* __restrict__ wgt,
    const float* __restrict__ bias, float* __restrict__ out,
    int H, int W, int Cout) {
    const int n    = blockIdx.y;
    const int tile = blockIdx.x;                       // 64 positions per tile
    const int tid  = threadIdx.x;
    const int lane = tid & 31;
    const int wv   = tid >> 5;
    const int HW   = H * W;
    const int m    = lane & 15;                        // A row / B col / C col
    const int g    = lane >> 4;                        // lane-half selector

    __shared__ __align__(16) __bf16 Aw[NMT * 16 * 448]; // weights, row-major K
    __shared__ __align__(16) __bf16 Bc[64 * 32];        // im2col chunk

    // stage weights once (zero-pad K 432..447 and rows >= Cout)
    const int atot = NMT * 16 * 448;
    for (int e = tid; e < atot; e += 128) {
        int row = e / 448, k = e - row * 448;
        float val = (row < Cout && k < 432) ? wgt[row * 432 + k] : 0.0f;
        Aw[e] = f2bf(val);
    }

    const v8f zero = {0.f, 0.f, 0.f, 0.f, 0.f, 0.f, 0.f, 0.f};
    v8f acc[NMT];
#pragma unroll
    for (int mt = 0; mt < NMT; ++mt) acc[mt] = zero;

    const int base_pos = tile * 64;
    const float* inN = in + (size_t)n * 48 * HW;

    for (int ks = 0; ks < 14; ++ks) {                  // 14 * 32 = 448 K
        __syncthreads();
        // stage B chunk: 64 positions x 32 K values (im2col gather)
#pragma unroll
        for (int ei = 0; ei < 16; ++ei) {
            int e = tid + ei * 128;
            int pos = e >> 5, kk = e & 31;
            int k = ks * 32 + kk;
            float val = 0.0f;
            if (k < 432) {
                int ci = k / 9, r = k - ci * 9;
                int ky = r / 3, kx = r - ky * 3;
                int pg = base_pos + pos;
                int y = pg / W, x = pg - y * W;
                int iy = y + ky - 1, ix = x + kx - 1;
                if (iy >= 0 && iy < H && ix >= 0 && ix < W)
                    val = inN[(size_t)ci * HW + iy * W + ix];
            }
            Bc[e] = f2bf(val);
        }
        __syncthreads();

        // B fragment (lane col = m, elements = K 16g..16g+15, contiguous order)
        const __bf16* brow = &Bc[((wv << 4) + m) * 32 + (g << 4)];
        v16bf bfr = cat16(*(const v8bf*)(brow), *(const v8bf*)(brow + 8));

#pragma unroll
        for (int mt = 0; mt < NMT; ++mt) {
            // A fragment: lane row m, elems = K 16g..16g+15
            const __bf16* arow = &Aw[((mt << 4) + m) * 448 + ks * 32];
            v16bf afr = cat16(*(const v8bf*)(arow + (g << 4)),
                              *(const v8bf*)(arow + 8 + (g << 4)));
            acc[mt] = __builtin_amdgcn_wmma_f32_16x16x32_bf16(
                false, afr, false, bfr, (short)0, acc[mt], false, false);
        }
    }

    // D layout: VGPR r -> row r + 8g, col = m
    int posg = base_pos + (wv << 4) + m;
#pragma unroll
    for (int mt = 0; mt < NMT; ++mt) {
#pragma unroll
        for (int r = 0; r < 8; ++r) {
            int co = (mt << 4) + r + (g << 3);
            if (co < Cout)
                out[((size_t)n * Cout + co) * HW + posg] = acc[mt][r] + bias[co];
        }
    }
}

// ---------------------------------------------------------------------------
// inorm -> prelu [-> inorm -> +residual]  per (n,c) channel.
// blockIdx.x = n*48+c ; 256 threads, HW/256 elems each held in registers.
// flags: 1 = second inorm + residual add; 0 = stop after prelu.
// ---------------------------------------------------------------------------
__global__ __launch_bounds__(256) void inorm_prelu_kernel(
    const float* __restrict__ conv, const float* __restrict__ resid,
    const float* __restrict__ a_ptr, float* __restrict__ out,
    int HW, int flags) {
    const int nc   = blockIdx.x;
    const int tid  = threadIdx.x;
    const int lane = tid & 31, wid = tid >> 5;
    const int cnt  = HW >> 8;                          // 16 @64x64, 4 @32x32
    const float a  = *a_ptr;
    const float invHW = 1.0f / (float)HW;

    __shared__ float redA[8], redB[8], bc[2];

    float v[16];
    float s1 = 0.f, s2 = 0.f;
    const float* src = conv + (size_t)nc * HW;
    for (int i = 0; i < cnt; ++i) {
        float t = src[tid + (i << 8)];
        v[i] = t; s1 += t; s2 += t * t;
    }
    for (int off = 16; off > 0; off >>= 1) {
        s1 += __shfl_down(s1, off, 32);
        s2 += __shfl_down(s2, off, 32);
    }
    if (lane == 0) { redA[wid] = s1; redB[wid] = s2; }
    __syncthreads();
    if (tid == 0) {
        float a1 = 0.f, a2 = 0.f;
        for (int i = 0; i < 8; ++i) { a1 += redA[i]; a2 += redB[i]; }
        float mean = a1 * invHW;
        float var  = a2 * invHW - mean * mean;
        bc[0] = mean; bc[1] = rsqrtf(var + 1e-5f);
    }
    __syncthreads();
    float mean = bc[0], inv = bc[1];

    s1 = 0.f; s2 = 0.f;
    for (int i = 0; i < cnt; ++i) {
        float t = (v[i] - mean) * inv;
        t = (t >= 0.f) ? t : a * t;                    // prelu
        v[i] = t; s1 += t; s2 += t * t;
    }

    float* o = out + (size_t)nc * HW;
    if (flags) {
        __syncthreads();
        for (int off = 16; off > 0; off >>= 1) {
            s1 += __shfl_down(s1, off, 32);
            s2 += __shfl_down(s2, off, 32);
        }
        if (lane == 0) { redA[wid] = s1; redB[wid] = s2; }
        __syncthreads();
        if (tid == 0) {
            float a1 = 0.f, a2 = 0.f;
            for (int i = 0; i < 8; ++i) { a1 += redA[i]; a2 += redB[i]; }
            float mean2 = a1 * invHW;
            float var2  = a2 * invHW - mean2 * mean2;
            bc[0] = mean2; bc[1] = rsqrtf(var2 + 1e-5f);
        }
        __syncthreads();
        float m2 = bc[0], i2 = bc[1];
        const float* rsd = resid + (size_t)nc * HW;
        for (int i = 0; i < cnt; ++i)
            o[tid + (i << 8)] = (v[i] - m2) * i2 + rsd[tid + (i << 8)];
    } else {
        for (int i = 0; i < cnt; ++i)
            o[tid + (i << 8)] = v[i];
    }
}

// ---------------------------------------------------------------------------
// Fused: bilinear x8 upsample of skernel -> softmax over 9 -> 3x3 weighted
// gather over reflect-padded x.  One block per (n, output row Y), 256 threads.
// ---------------------------------------------------------------------------
__global__ __launch_bounds__(256) void fuse_apply_kernel(
    const float* __restrict__ x, const float* __restrict__ sk,
    float* __restrict__ out) {
    const int Y = blockIdx.x, n = blockIdx.y, X = threadIdx.x;

    const float sy = (float)Y * (31.0f / 255.0f);
    int iy0 = (int)sy; int iy1 = min(iy0 + 1, 31); float fy = sy - (float)iy0;
    const float sx = (float)X * (31.0f / 255.0f);
    int ix0 = (int)sx; int ix1 = min(ix0 + 1, 31); float fx = sx - (float)ix0;

    float wk[9];
    float mx = -1e30f;
    const float* skn = sk + (size_t)n * 9 * 1024;
#pragma unroll
    for (int p = 0; p < 9; ++p) {
        const float* sp = skn + p * 1024;
        float v00 = sp[iy0 * 32 + ix0], v01 = sp[iy0 * 32 + ix1];
        float v10 = sp[iy1 * 32 + ix0], v11 = sp[iy1 * 32 + ix1];
        float v = (v00 * (1.f - fy) + v10 * fy) * (1.f - fx)
                + (v01 * (1.f - fy) + v11 * fy) * fx;
        wk[p] = v; mx = fmaxf(mx, v);
    }
    float s = 0.f;
#pragma unroll
    for (int p = 0; p < 9; ++p) { wk[p] = __expf(wk[p] - mx); s += wk[p]; }
    float invs = 1.f / s;
#pragma unroll
    for (int p = 0; p < 9; ++p) wk[p] *= invs;

    __shared__ float rows[3][258];
    int yy0 = refl(Y - 1, 256), yy1 = Y, yy2 = refl(Y + 1, 256);

    for (int c = 0; c < 48; ++c) {
        const float* xc = x + ((size_t)(n * 48 + c)) * 65536;
        __syncthreads();
        for (int j = threadIdx.x; j < 258; j += 256) {
            int xx = refl(j - 1, 256);
            rows[0][j] = xc[yy0 * 256 + xx];
            rows[1][j] = xc[yy1 * 256 + xx];
            rows[2][j] = xc[yy2 * 256 + xx];
        }
        __syncthreads();
        float acc = 0.f;
#pragma unroll
        for (int r = 0; r < 3; ++r)
#pragma unroll
            for (int d = 0; d < 3; ++d)
                acc += wk[r * 3 + d] * rows[r][X + d];
        out[((size_t)(n * 48 + c) * 256 + Y) * 256 + X] = acc;
    }
}

// ---------------------------------------------------------------------------
extern "C" void kernel_launch(void* const* d_in, const int* in_sizes, int n_in,
                              void* d_out, int out_size, void* d_ws, size_t ws_size,
                              hipStream_t stream) {
    (void)in_sizes; (void)n_in; (void)out_size; (void)ws_size;

    const float* x       = (const float*)d_in[0];
    const float* x_      = (const float*)d_in[1];
    const float* pre1_w  = (const float*)d_in[2];
    const float* pre1_b  = (const float*)d_in[3];
    const float* pre1_a  = (const float*)d_in[4];
    const float* pre2_w  = (const float*)d_in[5];
    const float* pre2_b  = (const float*)d_in[6];
    const float* pre2_a  = (const float*)d_in[7];
    const float* prek_w1 = (const float*)d_in[8];
    const float* prek_b1 = (const float*)d_in[9];
    const float* prek_a  = (const float*)d_in[10];
    const float* prek_w2 = (const float*)d_in[11];
    const float* prek_b2 = (const float*)d_in[12];
    float* out = (float*)d_out;

    // workspace carving (floats); total ~8.0 MB
    float* A64 = (float*)d_ws;          // (4,48,64,64)  786432
    float* B64 = A64 + 786432;          // (4,48,64,64)  786432
    float* C32 = B64 + 786432;          // (4,48,32,32)  196608
    float* D32 = C32 + 196608;          // (4,48,32,32)  196608
    float* SK  = D32 + 196608;          // (4, 9,32,32)   36864

    downsample4_kernel<<<3072, 256, 0, stream>>>(x_, A64);

    for (int i = 0; i < 3; ++i) {
        conv3x3_wmma<3><<<dim3(64, 4), 128, 0, stream>>>(
            A64, pre1_w + i * 20736, pre1_b + i * 48, B64, 64, 64, 48);
        inorm_prelu_kernel<<<192, 256, 0, stream>>>(
            B64, A64, pre1_a + i, A64, 4096, 1);
    }

    maxpool2_kernel<<<768, 256, 0, stream>>>(A64, C32);

    for (int i = 0; i < 3; ++i) {
        conv3x3_wmma<3><<<dim3(16, 4), 128, 0, stream>>>(
            C32, pre2_w + i * 20736, pre2_b + i * 48, D32, 32, 32, 48);
        inorm_prelu_kernel<<<192, 256, 0, stream>>>(
            D32, C32, pre2_a + i, C32, 1024, 1);
    }

    conv3x3_wmma<3><<<dim3(16, 4), 128, 0, stream>>>(
        C32, prek_w1, prek_b1, D32, 32, 32, 48);
    inorm_prelu_kernel<<<192, 256, 0, stream>>>(
        D32, nullptr, prek_a, D32, 1024, 0);

    conv3x3_wmma<1><<<dim3(16, 4), 128, 0, stream>>>(
        D32, prek_w2, prek_b2, SK, 32, 32, 9);

    fuse_apply_kernel<<<dim3(256, 4), 256, 0, stream>>>(x, SK, out);
}
